// MoE_64742337020148
// MI455X (gfx1250) — matrix operand order repaired and checked by reading, not compiled, not run on hardware
//
#include <hip/hip_runtime.h>
#include <math.h>

// ---------------- problem constants ----------------
#define BATCH    4096
#define NEXP     8
#define NPAIR    (2 * BATCH)      // top-2 dispatch pairs
#define EPS_LOG  2.2204460492503131e-16f

typedef __attribute__((ext_vector_type(16))) _Float16 v16h;
typedef __attribute__((ext_vector_type(8)))  _Float16 v8h;
typedef __attribute__((ext_vector_type(8)))  float    v8f;

// A-fragment = two contiguous 8-half chunks (contiguous k order, same as the B fragments):
// lane L (M=L%16): K = kb+{0..7} and kb+{8..15}, kb = (L>=16 ? 16 : 0).
union Afrag { v16h v; struct { v8h lo, hi; } p; };
union Row32 { v8h v[4]; _Float16 h[32]; };
union Row8  { v8h v;    _Float16 h[8];  };

// =====================================================================
// Kernel 1: gating.  logits = x_flat[4096,3072] @ w_gate[3072,8]; top-2
// softmax gates; importance/load accumulation.  One block per sample.
// =====================================================================
__global__ __launch_bounds__(256) void gate_kernel(
    const float* __restrict__ x, const float* __restrict__ w_gate,
    int* __restrict__ top2_idx, float* __restrict__ top2_gate,
    float* __restrict__ importance, float* __restrict__ loadv)
{
    __shared__ float red[256][NEXP];
    int b = blockIdx.x, t = threadIdx.x;
    const float* xb = x + (size_t)b * 3072;

    float acc[NEXP];
#pragma unroll
    for (int e = 0; e < NEXP; ++e) acc[e] = 0.f;

    for (int i = t; i < 3072; i += 256) {
        float xv = xb[i];
        const float* wr = w_gate + (size_t)i * NEXP;
#pragma unroll
        for (int e = 0; e < NEXP; ++e) acc[e] += xv * wr[e];
    }
#pragma unroll
    for (int e = 0; e < NEXP; ++e) red[t][e] = acc[e];
    __syncthreads();
    for (int s = 128; s > 0; s >>= 1) {
        if (t < s)
#pragma unroll
            for (int e = 0; e < NEXP; ++e) red[t][e] += red[t + s][e];
        __syncthreads();
    }
    if (t == 0) {
        float l[NEXP];
#pragma unroll
        for (int e = 0; e < NEXP; ++e) l[e] = red[0][e];
        int i0 = 0;
        for (int e = 1; e < NEXP; ++e) if (l[e] > l[i0]) i0 = e;
        int i1 = (i0 == 0) ? 1 : 0;
        for (int e = 0; e < NEXP; ++e) if (e != i0 && l[e] > l[i1]) i1 = e;
        float m  = fmaxf(l[i0], l[i1]);
        float e0 = expf(l[i0] - m), e1 = expf(l[i1] - m);   // TEMP = 1.0
        float inv = 1.f / (e0 + e1);
        float g0 = e0 * inv, g1 = e1 * inv;
        top2_idx[2 * b]     = i0;  top2_gate[2 * b]     = g0;
        top2_idx[2 * b + 1] = i1;  top2_gate[2 * b + 1] = g1;
        atomicAdd(&importance[i0], g0);
        atomicAdd(&importance[i1], g1);
        atomicAdd(&loadv[i0], 1.f);
        atomicAdd(&loadv[i1], 1.f);
    }
}

// =====================================================================
// Kernel 2: fused expert forward, one (sample, expert) pair per block.
// Implicit-GEMM convs on v_wmma_f32_16x16x32_f16 with cooperative,
// double-buffered im2col panels and b128 LDS fragment traffic.
// =====================================================================
struct PhaseA {                              // conv1 working set
    _Float16 panel[512][32] __attribute__((aligned(32)));    // im2col half-panel
    _Float16 w1B[32][32]    __attribute__((aligned(32)));    // [outch][k] (27 pad 32)
    float    a1[32], c1[32];                                 // fused BN scale/shift
    _Float16 c1out[32][32][32] __attribute__((aligned(32))); // conv1 ReLU out
};
struct PhaseB {                              // conv2 working set (overlaps A)
    _Float16 w2B[64][288] __attribute__((aligned(32)));      // [outch][k]
    float    a2[64], c2[64];
    _Float16 c2out[64][16][16]    __attribute__((aligned(32)));
    _Float16 panel2[2][256][32]   __attribute__((aligned(32))); // double-buffered
};
union PhaseU { PhaseA A; PhaseB Bp; };

__global__ __launch_bounds__(256) void expert_kernel(
    const float* __restrict__ x,
    const int*   __restrict__ top2_idx, const float* __restrict__ top2_gate,
    const float* __restrict__ w1, const float* __restrict__ b1,
    const float* __restrict__ g1, const float* __restrict__ be1,
    const float* __restrict__ rm1, const float* __restrict__ rv1,
    const float* __restrict__ w2, const float* __restrict__ b2,
    const float* __restrict__ g2, const float* __restrict__ be2,
    const float* __restrict__ rm2, const float* __restrict__ rv2,
    const float* __restrict__ fc_w, const float* __restrict__ fc_b,
    float* __restrict__ combined)
{
    __shared__ PhaseU u;
    __shared__ _Float16 h1[32][18][18];   // pooled conv1 (zero halo)
    __shared__ float    h2[4096];         // pooled conv2, torch-flatten order

    int pair = blockIdx.x;
    int b    = pair >> 1;
    int e    = top2_idx[pair];
    float gate = top2_gate[pair];
    int t = threadIdx.x, lane = t & 31, wv = t >> 5;
    int mrow  = lane & 15;                  // in-tile row/col index
    int abase = (lane & 16) ? 16 : 0;       // A-fragment K base (half-wave)
    int bbase = (lane & 16) ? 16 : 0;       // B-fragment K base (half-wave)

    const float* xb  = x + (size_t)b * 3072;
    const float* w2e = w2 + (size_t)e * 64 * 288;
    const float* fw  = fc_w + (size_t)e * 10 * 4096;

    // warm WGP$/L2 for phase-B weights while conv1 runs (global_prefetch_b8)
    for (int i = t * 32; i < 64 * 288; i += 256 * 32)
        __builtin_prefetch(w2e + i, 0, 3);

    // ---------- conv1 weights [outch][k] + BN consts ----------
    const float* w1e = w1 + (size_t)e * 32 * 27;        // [o][ci][ky][kx]
    for (int i = t; i < 32 * 32; i += 256) {
        int o = i >> 5, k = i & 31;
        u.A.w1B[o][k] = (k < 27) ? (_Float16)w1e[o * 27 + k] : (_Float16)0.f;
    }
    if (t < 32) {
        float s = g1[e * 32 + t] * rsqrtf(rv1[e * 32 + t] + 1e-5f);
        u.A.a1[t] = s;
        u.A.c1[t] = (b1[e * 32 + t] - rm1[e * 32 + t]) * s + be1[e * 32 + t];
    }
    __syncthreads();

    // conv1 B fragments: one v16h (2x ds_load_b128) per N-tile, K single step
    v16h b1f[2];
#pragma unroll
    for (int nt = 0; nt < 2; ++nt)
        b1f[nt] = *(const v16h*)&u.A.w1B[nt * 16 + mrow][bbase];

    // ---------- conv1: M=1024 positions in two 512-row panel halves ----------
    for (int half = 0; half < 2; ++half) {
        // cooperative im2col from global x (L2-resident): 2 rows per thread
        for (int m = t; m < 512; m += 256) {
            int p = half * 512 + m;
            int oy = p >> 5, ox = p & 31;
            Row32 row;
#pragma unroll
            for (int k = 0; k < 32; ++k) {
                float v = 0.f;
                if (k < 27) {
                    int ci = k / 9, rr = k % 9, ky = rr / 3, kx = rr % 3;
                    int iy = oy + ky - 1, ix = ox + kx - 1;
                    if (iy >= 0 && iy < 32 && ix >= 0 && ix < 32)
                        v = xb[ci * 1024 + iy * 32 + ix];
                }
                row.h[k] = (_Float16)v;
            }
#pragma unroll
            for (int q = 0; q < 4; ++q)
                *(v8h*)&u.A.panel[m][q * 8] = row.v[q];
        }
        __syncthreads();

        // 32 Mtiles x 2 Ntiles this half; wave wv owns Mtiles wv*4..wv*4+3
#pragma unroll
        for (int mi = 0; mi < 4; ++mi) {
            int mtl = wv * 4 + mi;
            Afrag fa;
            fa.p.lo = *(const v8h*)&u.A.panel[mtl * 16 + mrow][abase];
            fa.p.hi = *(const v8h*)&u.A.panel[mtl * 16 + mrow][abase + 8];
#pragma unroll
            for (int nt = 0; nt < 2; ++nt) {
                v8f c = {};
                v8f d = __builtin_amdgcn_wmma_f32_16x16x32_f16(
                            false, fa.v, false, b1f[nt], (short)0, c, false, false);
                int ch = nt * 16 + mrow;
                int m0 = half * 512 + mtl * 16 + ((lane & 16) ? 8 : 0);
                int y = m0 >> 5, x0 = m0 & 31;      // 8 consecutive x, 16B aligned
                float s = u.A.a1[ch], cc = u.A.c1[ch];
                Row8 o8;
#pragma unroll
                for (int r = 0; r < 8; ++r)
                    o8.h[r] = (_Float16)fmaxf(d[r] * s + cc, 0.f);
                *(v8h*)&u.A.c1out[ch][y][x0] = o8.v;
            }
        }
        __syncthreads();   // panel consumed before next half rebuild
    }

    // ---------- maxpool 2x2 -> h1 (zero halo for conv2 SAME padding) ----------
    for (int i = t; i < 32 * 18 * 18; i += 256) {
        int ch = i / 324, r = i % 324, y = r / 18, xx = r % 18;
        if (y == 0 || y == 17 || xx == 0 || xx == 17)
            h1[ch][y][xx] = (_Float16)0.f;
    }
    for (int i = t; i < 32 * 16 * 16; i += 256) {
        int ch = i >> 8, r = i & 255, py = r >> 4, px = r & 15;
        float v00 = (float)u.A.c1out[ch][2 * py][2 * px];
        float v01 = (float)u.A.c1out[ch][2 * py][2 * px + 1];
        float v10 = (float)u.A.c1out[ch][2 * py + 1][2 * px];
        float v11 = (float)u.A.c1out[ch][2 * py + 1][2 * px + 1];
        h1[ch][py + 1][px + 1] = (_Float16)fmaxf(fmaxf(v00, v01), fmaxf(v10, v11));
    }
    __syncthreads();   // c1out dead; safe to overwrite union with phase B

    // ---------- phase B loads: conv2 weights [outch][k] + BN consts ----------
    {
        _Float16* w2flat = &u.Bp.w2B[0][0];
        for (int i = t; i < 64 * 288; i += 256)
            w2flat[i] = (_Float16)w2e[i];               // coalesced convert-copy
    }
    if (t < 64) {
        float s = g2[e * 64 + t] * rsqrtf(rv2[e * 64 + t] + 1e-5f);
        u.Bp.a2[t] = s;
        u.Bp.c2[t] = (b2[e * 64 + t] - rm2[e * 64 + t]) * s + be2[e * 64 + t];
    }
    // warm caches for the FC weight stream consumed at the end
    for (int i = t * 32; i < 10 * 4096; i += 256 * 32)
        __builtin_prefetch(fw + i, 0, 3);

    // im2col row owned by this thread (exactly 256 rows)
    int p2y = t >> 4, p2x = t & 15;

    // prologue: build panel for k-step 0 into buffer 0
    {
        Row32 row;
#pragma unroll
        for (int kk = 0; kk < 32; ++kk) {
            int ci = kk / 9, rr = kk % 9, ky = rr / 3, kx = rr % 3;
            row.h[kk] = h1[ci][p2y + ky][p2x + kx];
        }
#pragma unroll
        for (int q = 0; q < 4; ++q)
            *(v8h*)&u.Bp.panel2[0][t][q * 8] = row.v[q];
    }

    // ---------- conv2: M=256, N=64, K=288 (9 k-steps, double-buffered) ----------
    // wave wv owns Mtiles {wv*2, wv*2+1} x all 4 Ntiles -> 8 persistent accs
    v8f acc[8];
    v8f zero = {};
#pragma unroll
    for (int jj = 0; jj < 8; ++jj) acc[jj] = zero;

    for (int ks = 0; ks < 9; ++ks) {
        __syncthreads();               // panel2[ks&1] ready; other buffer free
        int cur = ks & 1;

        // fragment loads from current panel (into registers first)
        Afrag fa[2];
#pragma unroll
        for (int mi = 0; mi < 2; ++mi) {
            int mtl = wv * 2 + mi;
            fa[mi].p.lo = *(const v8h*)&u.Bp.panel2[cur][mtl * 16 + mrow][abase];
            fa[mi].p.hi = *(const v8h*)&u.Bp.panel2[cur][mtl * 16 + mrow][abase + 8];
        }
        v16h fb[4];
#pragma unroll
        for (int nt = 0; nt < 4; ++nt)
            fb[nt] = *(const v16h*)&u.Bp.w2B[nt * 16 + mrow][ks * 32 + bbase];

        // build next k-step's panel into the other buffer (overlaps WMMAs)
        if (ks < 8) {
            Row32 row;
#pragma unroll
            for (int kk = 0; kk < 32; ++kk) {
                int k = (ks + 1) * 32 + kk;
                int ci = k / 9, rr = k % 9, ky = rr / 3, kx = rr % 3;
                row.h[kk] = h1[ci][p2y + ky][p2x + kx];
            }
#pragma unroll
            for (int q = 0; q < 4; ++q)
                *(v8h*)&u.Bp.panel2[1 - cur][t][q * 8] = row.v[q];
        }

#pragma unroll
        for (int jj = 0; jj < 8; ++jj)
            acc[jj] = __builtin_amdgcn_wmma_f32_16x16x32_f16(
                          false, fa[jj >> 2].v, false, fb[jj & 3],
                          (short)0, acc[jj], false, false);
    }

    // conv2 epilogue: BN + ReLU, one b128 store per (job, lane)
#pragma unroll
    for (int jj = 0; jj < 8; ++jj) {
        int mt = wv * 2 + (jj >> 2), nt = jj & 3;
        int ch = nt * 16 + mrow;
        int m0 = mt * 16 + ((lane & 16) ? 8 : 0);
        int y = m0 >> 4, x0 = m0 & 15;
        float s = u.Bp.a2[ch], cc = u.Bp.c2[ch];
        Row8 o8;
#pragma unroll
        for (int r = 0; r < 8; ++r)
            o8.h[r] = (_Float16)fmaxf(acc[jj][r] * s + cc, 0.f);
        *(v8h*)&u.Bp.c2out[ch][y][x0] = o8.v;
    }
    __syncthreads();

    // ---------- maxpool 2x2 -> h2 (flatten: [ch][py][px]) ----------
    for (int i = t; i < 64 * 8 * 8; i += 256) {
        int ch = i >> 6, r = i & 63, py = r >> 3, px = r & 7;
        float v00 = (float)u.Bp.c2out[ch][2 * py][2 * px];
        float v01 = (float)u.Bp.c2out[ch][2 * py][2 * px + 1];
        float v10 = (float)u.Bp.c2out[ch][2 * py + 1][2 * px];
        float v11 = (float)u.Bp.c2out[ch][2 * py + 1][2 * px + 1];
        h2[ch * 64 + py * 8 + px] = fmaxf(fmaxf(v00, v01), fmaxf(v10, v11));
    }
    __syncthreads();

    // ---------- FC 4096->10, exp, gate, global combine ----------
    for (int o = wv; o < 10; o += 8) {
        float p = 0.f;
        for (int i = lane; i < 4096; i += 32)
            p += h2[i] * fw[o * 4096 + i];
        for (int off = 16; off > 0; off >>= 1)
            p += __shfl_down(p, off, 32);
        if (lane == 0) {
            float outv = p + fc_b[e * 10 + o];
            atomicAdd(&combined[b * 10 + o], expf(outv) * gate);
        }
    }
}

// =====================================================================
// Kernel 3: y = log(combined==0 ? eps : combined); loss scalar at end.
// =====================================================================
__global__ void finalize_kernel(const float* __restrict__ combined,
                                const float* __restrict__ importance,
                                const float* __restrict__ loadv,
                                float* __restrict__ out)
{
    int i = blockIdx.x * blockDim.x + threadIdx.x;
    if (i < BATCH * 10) {
        float v = combined[i];
        out[i] = logf(v == 0.f ? EPS_LOG : v);
    }
    if (i == 0) {
        float mi = 0.f, ml = 0.f;
        for (int e = 0; e < NEXP; ++e) { mi += importance[e]; ml += loadv[e]; }
        mi *= (1.f / NEXP);  ml *= (1.f / NEXP);
        float vi = 0.f, vl = 0.f;
        for (int e = 0; e < NEXP; ++e) {
            float di = importance[e] - mi, dl = loadv[e] - ml;
            vi += di * di;  vl += dl * dl;
        }
        vi *= (1.f / (NEXP - 1));  vl *= (1.f / (NEXP - 1));   // unbiased var
        float loss = (vi / (mi * mi + 1e-10f) + vl / (ml * ml + 1e-10f)) * 0.01f;
        out[BATCH * 10] = loss;
    }
}

// =====================================================================
extern "C" void kernel_launch(void* const* d_in, const int* in_sizes, int n_in,
                              void* d_out, int out_size, void* d_ws, size_t ws_size,
                              hipStream_t stream)
{
    const float* x      = (const float*)d_in[0];
    // d_in[1] = index (unused by the reference math)
    const float* w_gate = (const float*)d_in[2];
    const float* w1  = (const float*)d_in[3];
    const float* b1  = (const float*)d_in[4];
    const float* g1  = (const float*)d_in[5];
    const float* be1 = (const float*)d_in[6];
    const float* rm1 = (const float*)d_in[7];
    const float* rv1 = (const float*)d_in[8];
    const float* w2  = (const float*)d_in[9];
    const float* b2  = (const float*)d_in[10];
    const float* g2  = (const float*)d_in[11];
    const float* be2 = (const float*)d_in[12];
    const float* rm2 = (const float*)d_in[13];
    const float* rv2 = (const float*)d_in[14];
    const float* fc_w = (const float*)d_in[15];
    const float* fc_b = (const float*)d_in[16];
    float* outp = (float*)d_out;

    // workspace layout
    char*  ws         = (char*)d_ws;
    int*   top2_idx   = (int*)ws;                          // NPAIR ints
    float* top2_gate  = (float*)(ws + (size_t)NPAIR * 4);  // NPAIR floats
    float* importance = (float*)(ws + (size_t)NPAIR * 8);  // 8
    float* loadv      = importance + NEXP;                 // 8
    float* combined   = loadv + NEXP;                      // BATCH*10

    // zero accumulators (importance + load + combined are contiguous)
    hipMemsetAsync(importance, 0, (size_t)(2 * NEXP + BATCH * 10) * sizeof(float), stream);

    gate_kernel<<<BATCH, 256, 0, stream>>>(x, w_gate, top2_idx, top2_gate,
                                           importance, loadv);
    expert_kernel<<<NPAIR, 256, 0, stream>>>(x, top2_idx, top2_gate,
                                             w1, b1, g1, be1, rm1, rv1,
                                             w2, b2, g2, be2, rm2, rv2,
                                             fc_w, fc_b, combined);
    finalize_kernel<<<(BATCH * 10 + 256) / 256, 256, 0, stream>>>(
        combined, importance, loadv, outp);
}
